// MambaEnhancedEncoderLayer_10634339025373
// MI455X (gfx1250) — matrix operand order repaired and checked by reading, not compiled, not run on hardware
//
#include <hip/hip_runtime.h>
#include <cstdint>
#include <cstddef>

// ---------------------------------------------------------------------------
// MI455X (gfx1250) implementation of the Mamba+Attention hybrid encoder layer.
// All large GEMMs go through v_wmma_f32_16x16x32_bf16 (bf16 operands, f32
// accum), register-blocked 16x64 per wave (A-fragment reuse across 4 n-tiles)
// with global_prefetch feeding the K-stream. Attention is flash-style (no LxL
// score tensor). The sequential selective scan is a register-resident VALU
// kernel (v_exp_f32) over 8192 independent channels.
// ---------------------------------------------------------------------------

typedef __bf16 bf16;
typedef __attribute__((ext_vector_type(16))) __bf16 v16bf;
typedef __attribute__((ext_vector_type(8)))  __bf16 bf16x8;
typedef __attribute__((ext_vector_type(8)))  float  v8f;

#define DEV static __device__ __forceinline__

constexpr int Bb   = 8;
constexpr int Tt   = 24;
constexpr int Ww   = 48;
constexpr int Dd   = 512;
constexpr int Hh   = 8;
constexpr int Ee   = 64;      // head dim
constexpr int DI   = 1024;    // mamba d_inner
constexpr int Ssn  = 64;      // d_state
constexpr int Rr   = 32;      // dt_rank
constexpr int KCc  = 4;       // conv kernel
constexpr int DFFc = 2048;
constexpr int Ll   = Tt * Ww;       // 1152
constexpr int Mrows = Bb * Ll;      // 9216
constexpr int XPN  = Rr + 2 * Ssn;  // 160

DEV float sigmoid_(float x) { return 1.f / (1.f + __expf(-x)); }
DEV float silu_(float x)    { return x * sigmoid_(x); }
DEV float gelu_(float x)    { return 0.5f * x * (1.f + erff(x * 0.70710678118654752f)); }
DEV float softplus_(float x){ return (x > 20.f) ? x : log1pf(__expf(x)); }

// -------------------------------- f32 -> bf16 ------------------------------
__global__ void f2b_kernel(const float* __restrict__ in, bf16* __restrict__ out, size_t n) {
    size_t i = (size_t)blockIdx.x * blockDim.x + threadIdx.x;
    if (i < n) out[i] = (bf16)in[i];
}

// ------------------------------ generic WMMA GEMM --------------------------
// C[M,N] = act(A[M,K] @ W[N,K]^T + bias) (+ res). A,W bf16 K-contiguous.
// One wave per 16x64 output strip (4 accumulators, A fragment reused across
// the 4 n-tiles); 4 waves per 128-thread block. ACT: 0 none, 1 gelu, 2 softplus.
template <int ACT, bool BIAS, bool RES, bool OUTF, bool OUTB>
__global__ __launch_bounds__(128)
void gemm_bf16_kernel(const bf16* __restrict__ A, int lda,
                      const bf16* __restrict__ Wt, int ldw,
                      const float* __restrict__ bias,
                      const float* __restrict__ res, int ldr,
                      float* __restrict__ Cf, int ldc,
                      bf16* __restrict__ Cb, int ldcb,
                      int N, int K) {
    const int lane = threadIdx.x & 31;
    const int wv   = threadIdx.x >> 5;
    const int r    = lane & 15;    // row-in-tile (A) / col-in-tile (B,C)
    const int hf   = lane >> 4;    // half-wave selector
    const int m0   = blockIdx.x * 16;
    const int n0   = (blockIdx.y * 4 + wv) * 64;
    if (n0 >= N) return;           // wave-uniform exit (EXEC stays all-ones)

    const bf16* arow = A + (size_t)(m0 + r) * lda;
    const bf16* wrow[4];
    bool valid[4];
#pragma unroll
    for (int et = 0; et < 4; ++et) {
        valid[et] = (n0 + et * 16) < N;        // wave-uniform
        wrow[et]  = Wt + (size_t)(n0 + et * 16 + r) * ldw;
    }

    v8f acc[4];
#pragma unroll
    for (int et = 0; et < 4; ++et)
#pragma unroll
        for (int j = 0; j < 8; ++j) acc[et][j] = 0.f;

    for (int kb = 0; kb < K; kb += 32) {
        // A 16x32 frag: lanes 0-15 hold K 0-15, lanes 16-31 hold K 16-31
        bf16x8 a0 = *(const bf16x8*)(arow + kb + hf * 16);
        bf16x8 a1 = *(const bf16x8*)(arow + kb + hf * 16 + 8);
        v16bf av;
#pragma unroll
        for (int i = 0; i < 8; ++i) { av[i] = a0[i]; av[i + 8] = a1[i]; }

        if (kb + 128 < K) {               // speculative prefetch, 2 lines ahead
            __builtin_prefetch(arow + kb + 128, 0, 1);
            __builtin_prefetch(wrow[0] + kb + 128, 0, 1);
        }

#pragma unroll
        for (int et = 0; et < 4; ++et) {
            if (!valid[et]) continue;     // wave-uniform: EXEC stays full
            // B 32x16 frag: lane holds 16 contiguous K for its column (= W row)
            bf16x8 b0 = *(const bf16x8*)(wrow[et] + kb + hf * 16);
            bf16x8 b1 = *(const bf16x8*)(wrow[et] + kb + hf * 16 + 8);
            v16bf bv;
#pragma unroll
            for (int i = 0; i < 8; ++i) { bv[i] = b0[i]; bv[i + 8] = b1[i]; }
            acc[et] = __builtin_amdgcn_wmma_f32_16x16x32_bf16(false, av, false, bv,
                                                              (short)0, acc[et], false, false);
        }
    }

#pragma unroll
    for (int et = 0; et < 4; ++et) {
        if (!valid[et]) continue;
        const int n  = n0 + et * 16 + r;
        const float bn = BIAS ? bias[n] : 0.f;
#pragma unroll
        for (int j = 0; j < 8; ++j) {
            const int m = m0 + hf * 8 + j;      // C layout: lane holds rows hf*8+j
            float v = acc[et][j] + bn;
            if constexpr (ACT == 1) v = gelu_(v);
            if constexpr (ACT == 2) v = softplus_(v);
            if constexpr (RES)  v += res[(size_t)m * ldr + n];
            if constexpr (OUTF) Cf[(size_t)m * ldc + n] = v;
            if constexpr (OUTB) Cb[(size_t)m * ldcb + n] = (bf16)v;
        }
    }
}

// ------------------------- V transpose for P@V B-fragments -----------------
// v_bf [B, L, D] -> vt [B*H, E, L]  (so each B-fragment lane reads contiguous L)
__global__ void transpose_v_kernel(const bf16* __restrict__ v, bf16* __restrict__ vt) {
    size_t i = (size_t)blockIdx.x * blockDim.x + threadIdx.x;
    if (i >= (size_t)Mrows * Dd) return;
    int l = (int)(i % Ll);
    int e = (int)((i / Ll) % Ee);
    int h = (int)((i / ((size_t)Ll * Ee)) % Hh);
    int b = (int)(i / ((size_t)Ll * Ee * Hh));
    vt[i] = v[(size_t)(b * Ll + l) * Dd + h * Ee + e];
}

// --------------------------- flash attention (WMMA) ------------------------
// One wave per 16-query tile per (b,h). Online softmax over 32-key steps.
// Score tile: 2 wmmas (K=64 head dim). P@V: P staged through LDS (C-layout ->
// A-layout), 4 wmmas per step (four 16-wide e tiles, K=32 keys).
__global__ __launch_bounds__(128)
void flash_attn_kernel(const bf16* __restrict__ q, const bf16* __restrict__ k,
                       const bf16* __restrict__ vt,
                       const unsigned char* __restrict__ mask,
                       bf16* __restrict__ out) {
    __shared__ bf16 plds[4][16][32];
    const int lane = threadIdx.x & 31;
    const int wv   = threadIdx.x >> 5;
    const int r    = lane & 15;
    const int hf   = lane >> 4;
    const int b    = blockIdx.x / Hh;
    const int h    = blockIdx.x % Hh;
    const int qt   = blockIdx.y * 4 + wv;   // 0..71

    // Q fragments for both 32-wide e chunks (loaded once)
    const bf16* qrow = q + (size_t)(b * Ll + qt * 16 + r) * Dd + h * Ee;
    v16bf qf0, qf1;
    {
        bf16x8 a0 = *(const bf16x8*)(qrow + hf * 16);
        bf16x8 a1 = *(const bf16x8*)(qrow + hf * 16 + 8);
        bf16x8 a2 = *(const bf16x8*)(qrow + 32 + hf * 16);
        bf16x8 a3 = *(const bf16x8*)(qrow + 32 + hf * 16 + 8);
#pragma unroll
        for (int i = 0; i < 8; ++i) {
            qf0[i] = a0[i]; qf0[i + 8] = a1[i];
            qf1[i] = a2[i]; qf1[i + 8] = a3[i];
        }
    }

    v8f oacc[4];
    float m8[8], l8[8];
#pragma unroll
    for (int e = 0; e < 4; ++e)
#pragma unroll
        for (int j = 0; j < 8; ++j) oacc[e][j] = 0.f;
#pragma unroll
    for (int j = 0; j < 8; ++j) { m8[j] = -1e30f; l8[j] = 0.f; }

    for (int kb = 0; kb < Ll; kb += 32) {
        // ---- two 16x16 score tiles (keys kb..kb+15 and kb+16..kb+31) ----
        v8f s0 = {0.f,0.f,0.f,0.f,0.f,0.f,0.f,0.f};
        v8f s1 = {0.f,0.f,0.f,0.f,0.f,0.f,0.f,0.f};
        const bf16* k0row = k + (size_t)(b * Ll + kb + r) * Dd + h * Ee;
        const bf16* k1row = k0row + (size_t)16 * Dd;
        {
            bf16x8 c0 = *(const bf16x8*)(k0row + hf * 16);
            bf16x8 c1 = *(const bf16x8*)(k0row + hf * 16 + 8);
            bf16x8 c2 = *(const bf16x8*)(k0row + 32 + hf * 16);
            bf16x8 c3 = *(const bf16x8*)(k0row + 32 + hf * 16 + 8);
            v16bf kf0, kf1;
#pragma unroll
            for (int i = 0; i < 8; ++i) { kf0[i]=c0[i]; kf0[i+8]=c1[i]; kf1[i]=c2[i]; kf1[i+8]=c3[i]; }
            s0 = __builtin_amdgcn_wmma_f32_16x16x32_bf16(false, qf0, false, kf0, (short)0, s0, false, false);
            s0 = __builtin_amdgcn_wmma_f32_16x16x32_bf16(false, qf1, false, kf1, (short)0, s0, false, false);
        }
        {
            bf16x8 c0 = *(const bf16x8*)(k1row + hf * 16);
            bf16x8 c1 = *(const bf16x8*)(k1row + hf * 16 + 8);
            bf16x8 c2 = *(const bf16x8*)(k1row + 32 + hf * 16);
            bf16x8 c3 = *(const bf16x8*)(k1row + 32 + hf * 16 + 8);
            v16bf kf0, kf1;
#pragma unroll
            for (int i = 0; i < 8; ++i) { kf0[i]=c0[i]; kf0[i+8]=c1[i]; kf1[i]=c2[i]; kf1[i+8]=c3[i]; }
            s1 = __builtin_amdgcn_wmma_f32_16x16x32_bf16(false, qf0, false, kf0, (short)0, s1, false, false);
            s1 = __builtin_amdgcn_wmma_f32_16x16x32_bf16(false, qf1, false, kf1, (short)0, s1, false, false);
        }

        // ---- online softmax over these 32 columns ----
        const bool mk0 = mask[b * Ll + kb + r] != 0;
        const bool mk1 = mask[b * Ll + kb + 16 + r] != 0;
        float sv0[8], sv1[8], al[8];
#pragma unroll
        for (int j = 0; j < 8; ++j) {
            sv0[j] = mk0 ? -1e30f : s0[j] * 0.125f;   // 1/sqrt(64)
            sv1[j] = mk1 ? -1e30f : s1[j] * 0.125f;
        }
#pragma unroll
        for (int j = 0; j < 8; ++j) {
            float t = fmaxf(sv0[j], sv1[j]);
            t = fmaxf(t, __shfl_xor(t, 1));
            t = fmaxf(t, __shfl_xor(t, 2));
            t = fmaxf(t, __shfl_xor(t, 4));
            t = fmaxf(t, __shfl_xor(t, 8));
            float nm = fmaxf(m8[j], t);
            al[j] = __expf(m8[j] - nm);
            m8[j] = nm;
        }
        float p0[8], p1[8];
#pragma unroll
        for (int j = 0; j < 8; ++j) {
            p0[j] = __expf(sv0[j] - m8[j]);
            p1[j] = __expf(sv1[j] - m8[j]);
            float t = p0[j] + p1[j];
            t += __shfl_xor(t, 1);
            t += __shfl_xor(t, 2);
            t += __shfl_xor(t, 4);
            t += __shfl_xor(t, 8);
            l8[j] = l8[j] * al[j] + t;
        }
#pragma unroll
        for (int e = 0; e < 4; ++e)
#pragma unroll
            for (int j = 0; j < 8; ++j) oacc[e][j] *= al[j];

        // ---- stage P (16x32, bf16) through LDS: C-layout -> A-layout ----
#pragma unroll
        for (int j = 0; j < 8; ++j) {
            plds[wv][hf * 8 + j][r]      = (bf16)p0[j];
            plds[wv][hf * 8 + j][16 + r] = (bf16)p1[j];
        }
        __syncthreads();
        v16bf pf;
        {
            bf16x8 pa = *(const bf16x8*)&plds[wv][r][hf * 16];
            bf16x8 pb = *(const bf16x8*)&plds[wv][r][hf * 16 + 8];
#pragma unroll
            for (int i = 0; i < 8; ++i) { pf[i] = pa[i]; pf[i + 8] = pb[i]; }
        }

        // ---- P @ V : four 16-wide e tiles, K = 32 keys ----
#pragma unroll
        for (int et = 0; et < 4; ++et) {
            const bf16* vrow = vt + ((size_t)((b * Hh + h) * Ee + et * 16 + r)) * Ll
                             + kb + hf * 16;
            bf16x8 v0 = *(const bf16x8*)vrow;
            bf16x8 v1 = *(const bf16x8*)(vrow + 8);
            v16bf bv;
#pragma unroll
            for (int i = 0; i < 8; ++i) { bv[i] = v0[i]; bv[i + 8] = v1[i]; }
            oacc[et] = __builtin_amdgcn_wmma_f32_16x16x32_bf16(false, pf, false, bv,
                                                               (short)0, oacc[et], false, false);
        }
        __syncthreads();
    }

    // ---- normalize and write [B,L,D] bf16 ----
#pragma unroll
    for (int j = 0; j < 8; ++j) {
        const float inv = 1.f / l8[j];
        const size_t row = (size_t)(b * Ll + qt * 16 + hf * 8 + j) * Dd + h * Ee;
#pragma unroll
        for (int et = 0; et < 4; ++et)
            out[row + et * 16 + r] = (bf16)(oacc[et][j] * inv);
    }
}

// ---------------------- causal depthwise conv + SiLU -----------------------
__global__ void conv_silu_kernel(const float* __restrict__ xz,     // [M, 2*DI]
                                 const float* __restrict__ cw,     // [DI, KC]
                                 const float* __restrict__ cb,
                                 float* __restrict__ uact, bf16* __restrict__ uact_b) {
    size_t i = (size_t)blockIdx.x * blockDim.x + threadIdx.x;
    if (i >= (size_t)Mrows * DI) return;
    int c = (int)(i % DI);
    int l = (int)((i / DI) % Ll);
    int b = (int)(i / ((size_t)DI * Ll));
    float s = cb[c];
#pragma unroll
    for (int kk = 0; kk < KCc; ++kk) {
        int ls = l - (KCc - 1) + kk;
        if (ls >= 0) s += xz[(size_t)(b * Ll + ls) * (2 * DI) + c] * cw[c * KCc + kk];
    }
    s = silu_(s);
    uact[i]   = s;
    uact_b[i] = (bf16)s;
}

// ----------------------------- selective scan ------------------------------
// thread = (b, channel c); 64 states in registers; B/C broadcast via LDS.
__global__ __launch_bounds__(256)
void ssm_scan_kernel(const float* __restrict__ delta,  // [M, DI]
                     const float* __restrict__ uact,   // [M, DI]
                     const float* __restrict__ xdbc,   // [M, 160]
                     const float* __restrict__ A_log,  // [DI, S]
                     float* __restrict__ y) {          // [M, DI]
    __shared__ float Bs[Ssn], Cs[Ssn];
    const int tid = threadIdx.x;
    const int c   = blockIdx.x * blockDim.x + tid;
    const int b   = blockIdx.y;

    float hst[Ssn], Ac[Ssn];
#pragma unroll
    for (int s = 0; s < Ssn; ++s) {
        Ac[s]  = -__expf(A_log[(size_t)c * Ssn + s]);
        hst[s] = 0.f;
    }
    for (int l = 0; l < Ll; ++l) {
        const size_t row = (size_t)(b * Ll + l);
        if (tid < Ssn)            Bs[tid]       = xdbc[row * XPN + Rr + tid];
        else if (tid < 2 * Ssn)   Cs[tid - Ssn] = xdbc[row * XPN + Rr + Ssn + (tid - Ssn)];
        __syncthreads();
        const float dt = delta[row * DI + c];
        const float du = dt * uact[row * DI + c];
        float acc = 0.f;
#pragma unroll
        for (int s = 0; s < Ssn; ++s) {
            hst[s] = hst[s] * __expf(dt * Ac[s]) + du * Bs[s];
            acc += hst[s] * Cs[s];
        }
        y[row * DI + c] = acc;
        __syncthreads();
    }
}

// ------------------- y = (y + u*D) * silu(z)  -> bf16 ----------------------
__global__ void y_gate_kernel(const float* __restrict__ y, const float* __restrict__ uact,
                              const float* __restrict__ Dssm, const float* __restrict__ xz,
                              bf16* __restrict__ y_b) {
    size_t i = (size_t)blockIdx.x * blockDim.x + threadIdx.x;
    if (i >= (size_t)Mrows * DI) return;
    int c = (int)(i % DI);
    size_t row = i / DI;
    float z = xz[row * (2 * DI) + DI + c];
    y_b[i] = (bf16)((y[i] + uact[i] * Dssm[c]) * silu_(z));
}

// ---------------------- residual combine + LN1 + LN2 -----------------------
DEV float blockReduceSum(float v, float* red) {
    v += __shfl_xor(v, 1);  v += __shfl_xor(v, 2);
    v += __shfl_xor(v, 4);  v += __shfl_xor(v, 8);  v += __shfl_xor(v, 16);
    const int wv = threadIdx.x >> 5;
    if ((threadIdx.x & 31) == 0) red[wv] = v;
    __syncthreads();
    float s = (threadIdx.x < 8) ? red[threadIdx.x] : 0.f;
    s += __shfl_xor(s, 1);  s += __shfl_xor(s, 2);  s += __shfl_xor(s, 4);
    if (threadIdx.x == 0) red[0] = s;
    __syncthreads();
    float out = red[0];
    __syncthreads();
    return out;
}

__global__ __launch_bounds__(256)
void combine_ln_kernel(const float* __restrict__ xf, const float* __restrict__ xa,
                       const float* __restrict__ xm,
                       const float* __restrict__ g1, const float* __restrict__ b1,
                       const float* __restrict__ g2, const float* __restrict__ b2,
                       float* __restrict__ hout, bf16* __restrict__ hn) {
    __shared__ float red[8];
    const int row = blockIdx.x;
    const int t   = threadIdx.x;
    const size_t base = (size_t)row * Dd;
    float v[2];
#pragma unroll
    for (int i = 0; i < 2; ++i) {
        int c = t + i * 256;
        v[i] = xf[base + c] + xa[base + c] + xm[base + c];
    }
    float mean = blockReduceSum(v[0] + v[1], red) * (1.f / Dd);
    float sq = 0.f;
#pragma unroll
    for (int i = 0; i < 2; ++i) { float d = v[i] - mean; sq += d * d; }
    float var = blockReduceSum(sq, red) * (1.f / Dd);
    float rs = rsqrtf(var + 1e-5f);
    float hv[2];
#pragma unroll
    for (int i = 0; i < 2; ++i) {
        int c = t + i * 256;
        hv[i] = (v[i] - mean) * rs * g1[c] + b1[c];
        hout[base + c] = hv[i];
    }
    float mean2 = blockReduceSum(hv[0] + hv[1], red) * (1.f / Dd);
    float sq2 = 0.f;
#pragma unroll
    for (int i = 0; i < 2; ++i) { float d = hv[i] - mean2; sq2 += d * d; }
    float var2 = blockReduceSum(sq2, red) * (1.f / Dd);
    float rs2 = rsqrtf(var2 + 1e-6f);
#pragma unroll
    for (int i = 0; i < 2; ++i) {
        int c = t + i * 256;
        hn[base + c] = (bf16)((hv[i] - mean2) * rs2 * g2[c] + b2[c]);
    }
}

// ===========================================================================
extern "C" void kernel_launch(void* const* d_in, const int* in_sizes, int n_in,
                              void* d_out, int out_size, void* d_ws, size_t ws_size,
                              hipStream_t stream) {
    const float* x        = (const float*)d_in[0];
    const unsigned char* mask = (const unsigned char*)d_in[1];
    const float* Wq = (const float*)d_in[2];   const float* bq = (const float*)d_in[3];
    const float* Wk = (const float*)d_in[4];   const float* bk = (const float*)d_in[5];
    const float* Wv = (const float*)d_in[6];   const float* bv = (const float*)d_in[7];
    const float* Wo = (const float*)d_in[8];   const float* bo = (const float*)d_in[9];
    const float* in_proj_w = (const float*)d_in[10];
    const float* conv_w    = (const float*)d_in[11];
    const float* conv_b    = (const float*)d_in[12];
    const float* x_proj_w  = (const float*)d_in[13];
    const float* dt_proj_w = (const float*)d_in[14];
    const float* dt_proj_b = (const float*)d_in[15];
    const float* A_log     = (const float*)d_in[16];
    const float* D_ssm     = (const float*)d_in[17];
    const float* out_proj_w= (const float*)d_in[18];
    const float* ln1_g = (const float*)d_in[19]; const float* ln1_b = (const float*)d_in[20];
    const float* ffn_w1 = (const float*)d_in[21]; const float* ffn_b1 = (const float*)d_in[22];
    const float* ffn_w2 = (const float*)d_in[23]; const float* ffn_b2 = (const float*)d_in[24];
    const float* ln2_g = (const float*)d_in[25]; const float* ln2_b = (const float*)d_in[26];

    char* ws = (char*)d_ws;
    size_t off = 0;
    auto alloc = [&](size_t bytes) -> void* {
        void* p = ws + off;
        off += (bytes + 255) & ~(size_t)255;
        return p;
    };

    // bf16 weights
    bf16* wq_b   = (bf16*)alloc((size_t)Dd * Dd * 2);
    bf16* wk_b   = (bf16*)alloc((size_t)Dd * Dd * 2);
    bf16* wv_b   = (bf16*)alloc((size_t)Dd * Dd * 2);
    bf16* wo_b   = (bf16*)alloc((size_t)Dd * Dd * 2);
    bf16* inp_b  = (bf16*)alloc((size_t)2 * DI * Dd * 2);
    bf16* xpj_b  = (bf16*)alloc((size_t)XPN * DI * 2);
    bf16* dtp_b  = (bf16*)alloc((size_t)DI * Rr * 2);
    bf16* outp_b = (bf16*)alloc((size_t)Dd * DI * 2);
    bf16* f1_b   = (bf16*)alloc((size_t)DFFc * Dd * 2);
    bf16* f2_b   = (bf16*)alloc((size_t)Dd * DFFc * 2);
    // activations
    bf16*  xf_b   = (bf16*)alloc((size_t)Mrows * Dd * 2);
    bf16*  q_b    = (bf16*)alloc((size_t)Mrows * Dd * 2);
    bf16*  k_b    = (bf16*)alloc((size_t)Mrows * Dd * 2);
    bf16*  v_b    = (bf16*)alloc((size_t)Mrows * Dd * 2);
    bf16*  vt_b   = (bf16*)alloc((size_t)Mrows * Dd * 2);
    bf16*  ao_b   = (bf16*)alloc((size_t)Mrows * Dd * 2);
    float* x_attn = (float*)alloc((size_t)Mrows * Dd * 4);
    float* xz     = (float*)alloc((size_t)Mrows * 2 * DI * 4);
    float* uact   = (float*)alloc((size_t)Mrows * DI * 4);
    bf16*  uact_b = (bf16*)alloc((size_t)Mrows * DI * 2);
    float* xdbc   = (float*)alloc((size_t)Mrows * XPN * 4);
    bf16*  xdbc_b = (bf16*)alloc((size_t)Mrows * XPN * 2);
    float* delta  = (float*)alloc((size_t)Mrows * DI * 4);
    float* yssm   = (float*)alloc((size_t)Mrows * DI * 4);
    bf16*  y_b    = (bf16*)alloc((size_t)Mrows * DI * 2);
    float* x_mam  = (float*)alloc((size_t)Mrows * Dd * 4);
    float* hbuf   = (float*)alloc((size_t)Mrows * Dd * 4);
    bf16*  hn_b   = (bf16*)alloc((size_t)Mrows * Dd * 2);
    bf16*  ff1_b  = (bf16*)alloc((size_t)Mrows * DFFc * 2);

    auto cvt = [&](const float* src, bf16* dst, size_t n) {
        f2b_kernel<<<dim3((unsigned)((n + 255) / 256)), 256, 0, stream>>>(src, dst, n);
    };
    cvt(Wq, wq_b, (size_t)Dd * Dd);
    cvt(Wk, wk_b, (size_t)Dd * Dd);
    cvt(Wv, wv_b, (size_t)Dd * Dd);
    cvt(Wo, wo_b, (size_t)Dd * Dd);
    cvt(in_proj_w,  inp_b,  (size_t)2 * DI * Dd);
    cvt(x_proj_w,   xpj_b,  (size_t)XPN * DI);
    cvt(dt_proj_w,  dtp_b,  (size_t)DI * Rr);
    cvt(out_proj_w, outp_b, (size_t)Dd * DI);
    cvt(ffn_w1, f1_b, (size_t)DFFc * Dd);
    cvt(ffn_w2, f2_b, (size_t)Dd * DFFc);
    cvt(x, xf_b, (size_t)Mrows * Dd);

    const dim3 blk(128);
    auto ggrid = [](int N) { return dim3(Mrows / 16, (N + 255) / 256); };

    // ---------------- attention ----------------
    gemm_bf16_kernel<0, true, false, false, true><<<ggrid(Dd), blk, 0, stream>>>(
        xf_b, Dd, wq_b, Dd, bq, nullptr, 0, nullptr, 0, q_b, Dd, Dd, Dd);
    gemm_bf16_kernel<0, true, false, false, true><<<ggrid(Dd), blk, 0, stream>>>(
        xf_b, Dd, wk_b, Dd, bk, nullptr, 0, nullptr, 0, k_b, Dd, Dd, Dd);
    gemm_bf16_kernel<0, true, false, false, true><<<ggrid(Dd), blk, 0, stream>>>(
        xf_b, Dd, wv_b, Dd, bv, nullptr, 0, nullptr, 0, v_b, Dd, Dd, Dd);
    transpose_v_kernel<<<dim3((unsigned)(((size_t)Mrows * Dd + 255) / 256)), 256, 0, stream>>>(v_b, vt_b);
    flash_attn_kernel<<<dim3(Bb * Hh, (Ll / 16) / 4), blk, 0, stream>>>(q_b, k_b, vt_b, mask, ao_b);
    gemm_bf16_kernel<0, true, false, true, false><<<ggrid(Dd), blk, 0, stream>>>(
        ao_b, Dd, wo_b, Dd, bo, nullptr, 0, x_attn, Dd, nullptr, 0, Dd, Dd);

    // ---------------- mamba ----------------
    gemm_bf16_kernel<0, false, false, true, false><<<ggrid(2 * DI), blk, 0, stream>>>(
        xf_b, Dd, inp_b, Dd, nullptr, nullptr, 0, xz, 2 * DI, nullptr, 0, 2 * DI, Dd);
    conv_silu_kernel<<<dim3((unsigned)(((size_t)Mrows * DI + 255) / 256)), 256, 0, stream>>>(
        xz, conv_w, conv_b, uact, uact_b);
    gemm_bf16_kernel<0, false, false, true, true><<<ggrid(XPN), blk, 0, stream>>>(
        uact_b, DI, xpj_b, DI, nullptr, nullptr, 0, xdbc, XPN, xdbc_b, XPN, XPN, DI);
    gemm_bf16_kernel<2, true, false, true, false><<<ggrid(DI), blk, 0, stream>>>(
        xdbc_b, XPN, dtp_b, Rr, dt_proj_b, nullptr, 0, delta, DI, nullptr, 0, DI, Rr);
    ssm_scan_kernel<<<dim3(DI / 256, Bb), 256, 0, stream>>>(delta, uact, xdbc, A_log, yssm);
    y_gate_kernel<<<dim3((unsigned)(((size_t)Mrows * DI + 255) / 256)), 256, 0, stream>>>(
        yssm, uact, D_ssm, xz, y_b);
    gemm_bf16_kernel<0, false, false, true, false><<<ggrid(Dd), blk, 0, stream>>>(
        y_b, DI, outp_b, DI, nullptr, nullptr, 0, x_mam, Dd, nullptr, 0, Dd, DI);

    // ---------------- combine + LN1 + LN2 ----------------
    combine_ln_kernel<<<dim3(Mrows), 256, 0, stream>>>(
        x, x_attn, x_mam, ln1_g, ln1_b, ln2_g, ln2_b, hbuf, hn_b);

    // ---------------- FFN ----------------
    gemm_bf16_kernel<1, true, false, false, true><<<ggrid(DFFc), blk, 0, stream>>>(
        hn_b, Dd, f1_b, Dd, ffn_b1, nullptr, 0, nullptr, 0, ff1_b, DFFc, DFFc, Dd);
    gemm_bf16_kernel<0, true, true, true, false><<<ggrid(Dd), blk, 0, stream>>>(
        ff1_b, DFFc, f2_b, DFFc, ffn_b2, hbuf, Dd, (float*)d_out, Dd, nullptr, 0, Dd, DFFc);

    (void)in_sizes; (void)n_in; (void)out_size; (void)ws_size;
}
